// KAN_22651657519766
// MI455X (gfx1250) — compile-verified, matrix operand order checked by reading, not run on hardware
//
#include <hip/hip_runtime.h>
#include <math.h>

typedef __attribute__((ext_vector_type(16))) _Float16 v16h;
typedef __attribute__((ext_vector_type(8)))  _Float16 v8h;
typedef __attribute__((ext_vector_type(8)))  float    v8f;
typedef int v4i __attribute__((vector_size(16)));

#define BATCH   8192
#define DIN     1024
#define DOUT    1024
#define COEFF   8
#define KSPL    (DIN * COEFF)   /* 8192 */
#define NLAYER  2

#define BM 128
#define BN 256
#define BK 32
#define LDT 40                  /* LDS row stride in halves: 80B, 16B aligned */
#define TILE_A (BM * LDT)       /* halves per A buffer */
#define TILE_B (BN * LDT)       /* halves per B buffer */

#if __has_builtin(__builtin_amdgcn_global_load_async_to_lds_b128)
#define HAVE_ASYNC_LDS 1
typedef __attribute__((address_space(1))) v4i gv4i_t;
typedef __attribute__((address_space(3))) v4i lv4i_t;
#else
#define HAVE_ASYNC_LDS 0
#endif

__device__ __forceinline__ void wait_async0() {
#if __has_builtin(__builtin_amdgcn_s_wait_asynccnt)
    __builtin_amdgcn_s_wait_asynccnt(0);
#else
    asm volatile("s_wait_asynccnt 0" ::: "memory");
#endif
}

// ---------------------------------------------------------------------------
// f32 -> f16 weight conversion (grid-stride)
// ---------------------------------------------------------------------------
__global__ void kan_cvt_f16(const float* __restrict__ src,
                            _Float16* __restrict__ dst, long long n) {
    long long i = (long long)blockIdx.x * blockDim.x + threadIdx.x;
    long long stride = (long long)gridDim.x * blockDim.x;
    for (; i < n; i += stride) dst[i] = (_Float16)src[i];
}

// ---------------------------------------------------------------------------
// Activation kernel: exact GELU -> Ab (f16), B-spline bases (order 3, uniform
// grid h=0.4, knots t_j = 0.4j - 2.2) -> As (f16, 8 per element)
// ---------------------------------------------------------------------------
__global__ void kan_act(const float* __restrict__ X,
                        _Float16* __restrict__ Ab,
                        _Float16* __restrict__ As) {
    long long idx = (long long)blockIdx.x * blockDim.x + threadIdx.x;
    if (idx >= (long long)BATCH * DIN) return;
    float x = X[idx];

    // exact GELU: 0.5*x*(1+erf(x/sqrt(2)))
    Ab[idx] = (_Float16)(0.5f * x * (1.0f + erff(x * 0.7071067811865475f)));

    // Cox-de Boor on uniform grid (all denominators = k*h, never 0)
    float b[11];
#pragma unroll
    for (int j = 0; j < 11; ++j) {
        float tj = 0.4f * (float)j - 2.2f;
        b[j] = (x >= tj && x < tj + 0.4f) ? 1.0f : 0.0f;
    }
#pragma unroll
    for (int k = 1; k <= 3; ++k) {
        float invkh = 1.0f / (0.4f * (float)k);
#pragma unroll
        for (int j = 0; j < 11; ++j) {
            if (j > 10 - k) break;
            float tj   = 0.4f * (float)j - 2.2f;
            float tjk1 = tj + 0.4f * (float)(k + 1);  // t[j+k+1]
            b[j] = (x - tj) * invkh * b[j] + (tjk1 - x) * invkh * b[j + 1];
        }
    }
    v8h o;
#pragma unroll
    for (int j = 0; j < 8; ++j) o[j] = (_Float16)b[j];
    *(v8h*)(As + idx * COEFF) = o;   // 16B aligned
}

// ---------------------------------------------------------------------------
// WMMA GEMM helpers
// ---------------------------------------------------------------------------
__device__ __forceinline__ v16h make_frag(const _Float16* p) {
    v8h lo = *(const v8h*)p;         // this lane's K 0..7 half-pattern
    v8h hi = *(const v8h*)(p + 16);  // +32 bytes -> K 16..23 pattern
    return __builtin_shufflevector(lo, hi, 0, 1, 2, 3, 4, 5, 6, 7,
                                   8, 9, 10, 11, 12, 13, 14, 15);
}

#if HAVE_ASYNC_LDS
// Async fill of one (A,B) LDS tile pair: 6 x b128 per thread, ASYNCcnt-tracked.
__device__ __forceinline__ void tile_fill_async(
        const _Float16* __restrict__ gA, long long lda,
        const _Float16* __restrict__ gB, long long ldb,
        int m0, int n0, int k0, int t,
        _Float16* smA, _Float16* smB) {
#pragma unroll
    for (int c = 0; c < 2; ++c) {                 // A: 128 rows x 4 segs
        int idx = t + c * 256;
        int row = idx >> 2, seg = (idx & 3) * 8;
        __builtin_amdgcn_global_load_async_to_lds_b128(
            (gv4i_t*)(gA + (long long)(m0 + row) * lda + k0 + seg),
            (lv4i_t*)(smA + row * LDT + seg), 0, 0);
    }
#pragma unroll
    for (int c = 0; c < 4; ++c) {                 // B: 256 rows x 4 segs
        int idx = t + c * 256;
        int row = idx >> 2, seg = (idx & 3) * 8;
        __builtin_amdgcn_global_load_async_to_lds_b128(
            (gv4i_t*)(gB + (long long)(n0 + row) * ldb + k0 + seg),
            (lv4i_t*)(smB + row * LDT + seg), 0, 0);
    }
}
#else
struct Stage { v8h a[2]; v8h b[4]; };

__device__ __forceinline__ void gload(const _Float16* __restrict__ gA, long long lda,
                                      const _Float16* __restrict__ gB, long long ldb,
                                      int m0, int n0, int k0, int t, Stage& s) {
#pragma unroll
    for (int c = 0; c < 2; ++c) {
        int idx = t + c * 256;
        int row = idx >> 2, seg = (idx & 3) * 8;
        s.a[c] = *(const v8h*)(gA + (long long)(m0 + row) * lda + k0 + seg);
    }
#pragma unroll
    for (int c = 0; c < 4; ++c) {
        int idx = t + c * 256;
        int row = idx >> 2, seg = (idx & 3) * 8;
        s.b[c] = *(const v8h*)(gB + (long long)(n0 + row) * ldb + k0 + seg);
    }
}

__device__ __forceinline__ void sstore(_Float16* smA, _Float16* smB, int t,
                                       const Stage& s) {
#pragma unroll
    for (int c = 0; c < 2; ++c) {
        int idx = t + c * 256;
        int row = idx >> 2, seg = (idx & 3) * 8;
        *(v8h*)(smA + row * LDT + seg) = s.a[c];
    }
#pragma unroll
    for (int c = 0; c < 4; ++c) {
        int idx = t + c * 256;
        int row = idx >> 2, seg = (idx & 3) * 8;
        *(v8h*)(smB + row * LDT + seg) = s.b[c];
    }
}
#endif

// 64x64 wave tile: 4 A-frags, 4 B-frags, 16 WMMA per BK=32 step
__device__ __forceinline__ void mma_tile(const _Float16* smA, const _Float16* smB,
                                         int lane, int wm, int wn, v8f acc[4][4]) {
    int r  = lane & 15;
    int hb = (lane >> 4) * 8;   // half-offset select per ISA 16-bit A/B layout
    v16h a[4], b[4];
#pragma unroll
    for (int i = 0; i < 4; ++i)
        a[i] = make_frag(smA + (wm * 64 + i * 16 + r) * LDT + hb);
#pragma unroll
    for (int j = 0; j < 4; ++j)
        b[j] = make_frag(smB + (wn * 64 + j * 16 + r) * LDT + hb);
#pragma unroll
    for (int i = 0; i < 4; ++i)
#pragma unroll
        for (int j = 0; j < 4; ++j)
            acc[i][j] = __builtin_amdgcn_wmma_f32_16x16x32_f16(
                false, a[i], false, b[j], (short)0, acc[i][j], false, false);
}

__device__ __forceinline__ void gemm_phase(const _Float16* __restrict__ gA, long long lda,
                                           const _Float16* __restrict__ gB, long long ldb,
                                           int K, int m0, int n0,
                                           _Float16* smA, _Float16* smB,
                                           int t, int lane, int wm, int wn,
                                           v8f acc[4][4]) {
    const int nt = K / BK;
#if HAVE_ASYNC_LDS
    tile_fill_async(gA, lda, gB, ldb, m0, n0, 0, t, smA, smB);
    wait_async0();
    __syncthreads();
    for (int kt = 0; kt < nt; ++kt) {
        int cur = kt & 1, nxt = cur ^ 1;
        if (kt + 1 < nt)
            tile_fill_async(gA, lda, gB, ldb, m0, n0, (kt + 1) * BK, t,
                            smA + nxt * TILE_A, smB + nxt * TILE_B);
        mma_tile(smA + cur * TILE_A, smB + cur * TILE_B, lane, wm, wn, acc);
        wait_async0();
        __syncthreads();
    }
#else
    Stage s;
    gload(gA, lda, gB, ldb, m0, n0, 0, t, s);
    sstore(smA, smB, t, s);
    __syncthreads();
    for (int kt = 0; kt < nt; ++kt) {
        int cur = kt & 1, nxt = cur ^ 1;
        if (kt + 1 < nt) gload(gA, lda, gB, ldb, m0, n0, (kt + 1) * BK, t, s);
        mma_tile(smA + cur * TILE_A, smB + cur * TILE_B, lane, wm, wn, acc);
        if (kt + 1 < nt) sstore(smA + nxt * TILE_A, smB + nxt * TILE_B, t, s);
        __syncthreads();
    }
#endif
}

// ---------------------------------------------------------------------------
// Fused GEMM: H = Ab @ Wb^T + As @ Ws^T   (K = 1024 then K = 8192 phases)
// Workgroup tile 128x256, 8 waves as 2(M) x 4(N), each wave 64x64.
// ---------------------------------------------------------------------------
__global__ __launch_bounds__(256) void kan_gemm(
    const _Float16* __restrict__ Ab, const _Float16* __restrict__ Wb,
    const _Float16* __restrict__ As, const _Float16* __restrict__ Ws,
    float* __restrict__ H) {
    __shared__ __align__(16) _Float16 smA[2 * TILE_A];
    __shared__ __align__(16) _Float16 smB[2 * TILE_B];

    const int t    = threadIdx.x;
    const int lane = t & 31;
    const int w    = t >> 5;          // 8 waves
    const int wm   = w >> 2;          // 0..1 : 64-row slab
    const int wn   = w & 3;           // 0..3 : 64-col slab
    const int m0   = blockIdx.y * BM;
    const int n0   = blockIdx.x * BN;

    v8f acc[4][4];
#pragma unroll
    for (int i = 0; i < 4; ++i)
#pragma unroll
        for (int j = 0; j < 4; ++j) acc[i][j] = (v8f)(0.0f);

    gemm_phase(Ab, DIN,  Wb, DIN,  DIN,  m0, n0, smA, smB, t, lane, wm, wn, acc);
    gemm_phase(As, KSPL, Ws, KSPL, KSPL, m0, n0, smA, smB, t, lane, wm, wn, acc);

    // C/D layout: lanes 0-15 -> M = vgpr(0..7); lanes 16-31 -> M = 8+vgpr
    const int r   = lane & 15;
    const int sel = (lane >> 4) * 8;
#pragma unroll
    for (int i = 0; i < 4; ++i) {
#pragma unroll
        for (int j = 0; j < 4; ++j) {
            int col = n0 + wn * 64 + j * 16 + r;
#pragma unroll
            for (int g = 0; g < 8; ++g) {
                int row = m0 + wm * 64 + i * 16 + sel + g;
                H[(long long)row * DOUT + col] = acc[i][j][g];
            }
        }
    }
}

// ---------------------------------------------------------------------------
// LayerNorm + PReLU epilogue (one block per row)
// ---------------------------------------------------------------------------
__global__ void kan_ln_prelu(const float* __restrict__ H,
                             const float* __restrict__ ln_g,
                             const float* __restrict__ ln_b,
                             const float* __restrict__ prelu_a,
                             int layer, float* __restrict__ out) {
    const int row = blockIdx.x;
    const float* h = H + (long long)row * DOUT;
    float s = 0.0f, s2 = 0.0f;
    for (int c = threadIdx.x; c < DOUT; c += blockDim.x) {
        float v = h[c];
        s += v; s2 += v * v;
    }
    __shared__ float r1[256], r2[256];
    r1[threadIdx.x] = s; r2[threadIdx.x] = s2;
    __syncthreads();
    for (int st = 128; st > 0; st >>= 1) {
        if (threadIdx.x < st) {
            r1[threadIdx.x] += r1[threadIdx.x + st];
            r2[threadIdx.x] += r2[threadIdx.x + st];
        }
        __syncthreads();
    }
    float mu  = r1[0] * (1.0f / DOUT);
    float var = r2[0] * (1.0f / DOUT) - mu * mu;
    float inv = rsqrtf(var + 1e-5f);
    float alpha = prelu_a[layer];
    const float* gg = ln_g + (long long)layer * DOUT;
    const float* bb = ln_b + (long long)layer * DOUT;
    for (int c = threadIdx.x; c < DOUT; c += blockDim.x) {
        float v = (h[c] - mu) * inv * gg[c] + bb[c];
        out[(long long)row * DOUT + c] = (v >= 0.0f) ? v : alpha * v;
    }
}

// ---------------------------------------------------------------------------
// Host driver
// ---------------------------------------------------------------------------
extern "C" void kernel_launch(void* const* d_in, const int* in_sizes, int n_in,
                              void* d_out, int out_size, void* d_ws, size_t ws_size,
                              hipStream_t stream) {
    const float* x        = (const float*)d_in[0];
    const float* base_w   = (const float*)d_in[1];
    const float* spline_w = (const float*)d_in[2];
    const float* ln_g     = (const float*)d_in[3];
    const float* ln_b     = (const float*)d_in[4];
    const float* prelu_a  = (const float*)d_in[5];
    // d_in[6] = grid (uniform; recomputed analytically on device)

    char* ws = (char*)d_ws;
    _Float16* Ab   = (_Float16*)(ws);                          //  16 MiB
    _Float16* As   = (_Float16*)(ws + ((size_t)16  << 20));    // 128 MiB
    _Float16* Wb16 = (_Float16*)(ws + ((size_t)144 << 20));    //   4 MiB
    _Float16* Ws16 = (_Float16*)(ws + ((size_t)148 << 20));    //  32 MiB
    float*    Hbuf = (float*)   (ws + ((size_t)180 << 20));    //  32 MiB
    float*    Xbuf = (float*)   (ws + ((size_t)212 << 20));    //  32 MiB

    kan_cvt_f16<<<2048, 256, 0, stream>>>(base_w, Wb16,
        (long long)NLAYER * DOUT * DIN);
    kan_cvt_f16<<<8192, 256, 0, stream>>>(spline_w, Ws16,
        (long long)NLAYER * DOUT * KSPL);

    const float* xin = x;
    for (int l = 0; l < NLAYER; ++l) {
        kan_act<<<(BATCH * DIN) / 256, 256, 0, stream>>>(xin, Ab, As);

        dim3 gg(DOUT / BN, BATCH / BM);   // (4, 64)
        kan_gemm<<<gg, 256, 0, stream>>>(Ab,
                                         Wb16 + (size_t)l * DOUT * DIN,
                                         As,
                                         Ws16 + (size_t)l * DOUT * KSPL,
                                         Hbuf);

        float* outp = (l == NLAYER - 1) ? (float*)d_out : Xbuf;
        kan_ln_prelu<<<BATCH, 256, 0, stream>>>(Hbuf, ln_g, ln_b, prelu_a, l, outp);
        xin = Xbuf;
    }
}
